// GraphGeoModule_38998303048499
// MI455X (gfx1250) — matrix operand order repaired and checked by reading, not compiled, not run on hardware
//
#include <hip/hip_runtime.h>
#include <hip/hip_bf16.h>

// ---------------- problem constants ----------------
#define N_NODES 50000
#define N_EDGES 400000
#define D_IN    64
#define D_H     256
#define D_OUT   1024
#define N_IDS   8192

typedef __attribute__((ext_vector_type(16))) __bf16 v16bf;
typedef __attribute__((ext_vector_type(8)))  float  v8f;
typedef unsigned int u32x4 __attribute__((ext_vector_type(4)));
typedef int          i32x4 __attribute__((ext_vector_type(4)));
typedef int          i32x8 __attribute__((ext_vector_type(8)));

// ---------------- small helper kernels ----------------

__global__ void deg_init_kernel(float* __restrict__ deg) {
    int i = blockIdx.x * blockDim.x + threadIdx.x;
    if (i < N_NODES) deg[i] = 1.0f;               // self-loop contributes 1
}

__global__ void deg_count_kernel(const int* __restrict__ col, float* __restrict__ deg) {
    int e = blockIdx.x * blockDim.x + threadIdx.x;
    if (e < N_EDGES) atomicAdd(&deg[col[e]], 1.0f);
}

__global__ void deg_rsqrt_kernel(float* __restrict__ deg) {
    int i = blockIdx.x * blockDim.x + threadIdx.x;
    if (i < N_NODES) deg[i] = rsqrtf(deg[i]);     // deg >= 1 always
}

__global__ void mask_zero_kernel(int* __restrict__ mask) {
    int i = blockIdx.x * blockDim.x + threadIdx.x;
    if (i < N_NODES) mask[i] = 0;
}

__global__ void mask_set_kernel(const int* __restrict__ geo, int* __restrict__ mask) {
    int i = blockIdx.x * blockDim.x + threadIdx.x;
    if (i < N_IDS) mask[geo[i]] = 1;
}

// agg1[i] = dinv[i]^2 * X[i]  (self-loop term doubles as init; no memset needed)
__global__ void agg1_init_kernel(const float* __restrict__ X, const float* __restrict__ dinv,
                                 float* __restrict__ agg1) {
    int i = blockIdx.x * blockDim.x + threadIdx.x;   // over N_NODES*D_IN
    if (i < N_NODES * D_IN) {
        float d = dinv[i >> 6];
        agg1[i] = d * d * X[i];
    }
}

// scatter: agg1[c] += dinv[r]*dinv[c] * X[r]   (one block per edge, 64 features)
__global__ void agg1_edge_kernel(const int* __restrict__ row, const int* __restrict__ col,
                                 const float* __restrict__ dinv, const float* __restrict__ X,
                                 float* __restrict__ agg1) {
    int e = blockIdx.x;
    int f = threadIdx.x;
    int r = row[e], c = col[e];
    float nrm = dinv[r] * dinv[c];
    atomicAdd(&agg1[c * D_IN + f], nrm * X[r * D_IN + f]);
}

// agg2[i] = dinv[i]^2 * h1[i] for masked (geo) nodes only — other rows are never read
__global__ void agg2_init_kernel(const __bf16* __restrict__ h1, const float* __restrict__ dinv,
                                 const int* __restrict__ mask, float* __restrict__ agg2) {
    int i = blockIdx.x * blockDim.x + threadIdx.x;   // over N_NODES*D_H
    if (i >= N_NODES * D_H) return;
    int node = i >> 8;
    if (!mask[node]) return;
    float d = dinv[node];
    agg2[i] = d * d * (float)h1[i];
}

// scatter only into rows read by the final gather
__global__ void agg2_edge_kernel(const int* __restrict__ row, const int* __restrict__ col,
                                 const float* __restrict__ dinv, const int* __restrict__ mask,
                                 const __bf16* __restrict__ h1, float* __restrict__ agg2) {
    int e = blockIdx.x;
    int c = col[e];
    if (!mask[c]) return;                            // ~84% of blocks exit immediately
    int r = row[e];
    int f = threadIdx.x;
    float nrm = dinv[r] * dinv[c];
    atomicAdd(&agg2[c * D_H + f], nrm * (float)h1[r * D_H + f]);
}

// ---------------- weight pre-pack ----------------
// Pack row-major f32 W[K x N] into bf16 fragment-ordered P:
//   P[((n_tile * (K/32) + kk) * 32 + lane) * 16 + elem]
// so a wave's B fragment (32 K x 16 N, k = 16*half + elem order) is one
// contiguous 32-byte chunk per lane, and a 4-n-tile "n-group" slab is one
// contiguous 32 KB region (TDM-friendly).
__global__ void pack_w_kernel(const float* __restrict__ W, __bf16* __restrict__ P,
                              int K, int N) {
    const int kks = K >> 5;                // K/32 chunks
    const int total = (N >> 4) * kks * 32; // n_tiles * kks * 32 lanes
    int t = blockIdx.x * blockDim.x + threadIdx.x;
    if (t >= total) return;
    int lane   = t & 31;
    int kk     = (t >> 5) % kks;
    int n_tile = t / (32 * kks);
    int half   = lane >> 4;
    int n_col  = n_tile * 16 + (lane & 15);
    __bf16* dst = P + (size_t)t * 16;
#pragma unroll
    for (int p = 0; p < 8; ++p) {
        int k = kk * 32 + (half << 4) + (p << 1);   // B 32x16 layout: K pair per VGPR
        dst[2 * p]     = (__bf16)W[(size_t)k * N + n_col];
        dst[2 * p + 1] = (__bf16)W[(size_t)(k + 1) * N + n_col];
    }
}

// ---------------- WMMA GEMM kernels ----------------
// Load one A 16x32 bf16 fragment for this lane: two contiguous 8-float runs.
__device__ __forceinline__ v16bf load_a_frag(const float* __restrict__ arow,
                                             int kOff, int half) {
    const float* p0 = arow + kOff + (half << 4);         // K = half*16 .. +7
    const float* p1 = arow + kOff + (half << 4) + 8;     // K = half*16+8 .. +7
    float4 f0 = *(const float4*)(p0);
    float4 f1 = *(const float4*)(p0 + 4);
    float4 f2 = *(const float4*)(p1);
    float4 f3 = *(const float4*)(p1 + 4);
    v16bf a;
    a[0]  = (__bf16)f0.x; a[1]  = (__bf16)f0.y; a[2]  = (__bf16)f0.z; a[3]  = (__bf16)f0.w;
    a[4]  = (__bf16)f1.x; a[5]  = (__bf16)f1.y; a[6]  = (__bf16)f1.z; a[7]  = (__bf16)f1.w;
    a[8]  = (__bf16)f2.x; a[9]  = (__bf16)f2.y; a[10] = (__bf16)f2.z; a[11] = (__bf16)f2.w;
    a[12] = (__bf16)f3.x; a[13] = (__bf16)f3.y; a[14] = (__bf16)f3.z; a[15] = (__bf16)f3.w;
    return a;
}

// h1 = relu(agg1[50000x64] @ W1[64x256] + b1) -> bf16.
// Wave computes 16 rows x 64 cols (4 n-tiles); K=64 -> 2 K-steps, 4 WMMAs each.
// W1p is 32 KB total (L2-trivial): global path, no LDS staging.
__global__ void gemm1_wmma_kernel(const float* __restrict__ A, const __bf16* __restrict__ Wp,
                                  const float* __restrict__ bias, __bf16* __restrict__ H) {
    const int wave = threadIdx.x >> 5;
    const int lane = threadIdx.x & 31;
    const int m_tile  = blockIdx.x * 2 + (wave >> 2);
    const int n_group = wave & 3;                    // 4 n-tiles per wave
    if (m_tile >= N_NODES / 16) return;              // wave-uniform guard

    const int half = lane >> 4;
    const int mr   = lane & 15;
    const int rowM = m_tile * 16 + mr;
    const float* arow = A + (size_t)rowM * D_IN;

    const int kks = D_IN / 32;                       // 2
    v8f c0 = {}, c1 = {}, c2 = {}, c3 = {};
#pragma unroll
    for (int kk = 0; kk < kks; ++kk) {
        v16bf a = load_a_frag(arow, kk * 32, half);
        const __bf16* bp =
            Wp + ((size_t)((n_group * 4) * kks + kk) * 32 + lane) * 16;
        const size_t bstride = (size_t)kks * 32 * 16; // next n_tile, same kk
        v16bf b0 = *(const v16bf*)(bp);
        v16bf b1 = *(const v16bf*)(bp + bstride);
        v16bf b2 = *(const v16bf*)(bp + 2 * bstride);
        v16bf b3 = *(const v16bf*)(bp + 3 * bstride);
        c0 = __builtin_amdgcn_wmma_f32_16x16x32_bf16(false, a, false, b0, (short)0, c0, false, false);
        c1 = __builtin_amdgcn_wmma_f32_16x16x32_bf16(false, a, false, b1, (short)0, c1, false, false);
        c2 = __builtin_amdgcn_wmma_f32_16x16x32_bf16(false, a, false, b2, (short)0, c2, false, false);
        c3 = __builtin_amdgcn_wmma_f32_16x16x32_bf16(false, a, false, b3, (short)0, c3, false, false);
    }

    v8f acc[4] = {c0, c1, c2, c3};
#pragma unroll
    for (int j = 0; j < 4; ++j) {
        int n_col = (n_group * 4 + j) * 16 + mr;
        float bb = bias[n_col];
#pragma unroll
        for (int v = 0; v < 8; ++v) {                // C/D: M = v + 8*half
            int r = m_tile * 16 + (half << 3) + v;
            float val = acc[j][v] + bb;
            val = val > 0.0f ? val : 0.0f;
            H[(size_t)r * D_H + n_col] = (__bf16)val;
        }
    }
}

// out[8192x1024] = relu(agg2[geo_ids] @ W2[256x1024] + b2) -> bf16.
// Block = 8 waves = 8 m-tiles x 1 n-group (4 n-tiles). The 32 KB packed-B
// slab for the n-group is staged into LDS once per block via the Tensor Data
// Mover, then all 8 waves read B fragments with ds_load_b128.
// Grid: 64 m-blocks * 16 n-groups = 1024 blocks.
#define G2_KKS   (D_H / 32)                          // 8 K-steps
#define G2_SLAB  (4 * G2_KKS * 32 * 16)              // 16384 bf16 = 32 KB
__global__ __launch_bounds__(256) void gemm2_wmma_kernel(
        const float* __restrict__ A2, const int* __restrict__ geo,
        const __bf16* __restrict__ Wp, const float* __restrict__ bias,
        __bf16* __restrict__ out) {
    __shared__ __bf16 ldsB[G2_SLAB];

    const int wave = threadIdx.x >> 5;
    const int lane = threadIdx.x & 31;
    const int n_group = blockIdx.x & 15;
    const int m_tile  = (blockIdx.x >> 4) * 8 + wave;

    // ---- TDM stage: one descriptor, issued by wave 0 only (EXEC ignored) ----
    if (wave == 0) {
        const unsigned long long ga =
            (unsigned long long)(const void*)(Wp + (size_t)(n_group * 4) * G2_KKS * 32 * 16);
        const unsigned lds_addr = (unsigned)(size_t)(__bf16*)ldsB;

        u32x4 g0;
        g0[0] = 1u;                                   // count=1, user descriptor
        g0[1] = lds_addr;                             // LDS byte address
        g0[2] = (unsigned)(ga & 0xFFFFFFFFu);         // global_addr[31:0]
        g0[3] = (unsigned)((ga >> 32) & 0x01FFFFFFu)  // global_addr[56:32]
              | (2u << 30);                           // type=2 ("image")

        i32x8 g1;
        g1[0] = (int)(3u << 16);                      // data_size=8B; no multicast
        g1[1] = (int)(0x1000u << 16);                 // tensor_dim0 = 4096 (8B units)
        g1[2] = (int)(1u << 16);                      // tensor_dim1 = 1
        g1[3] = (int)(0x1000u << 16);                 // tile_dim0 = 4096 -> 32 KB line
        g1[4] = 0;                                    // tile_dim1/2 unused (1-D tile)
        g1[5] = 4096;                                 // tensor_dim0_stride
        g1[6] = 0;
        g1[7] = 0;

        i32x4 z4 = {0, 0, 0, 0};
#if __clang_major__ >= 23
        i32x8 z8 = {0, 0, 0, 0, 0, 0, 0, 0};
        __builtin_amdgcn_tensor_load_to_lds(g0, g1, z4, z4, z8, 0);
#else
        __builtin_amdgcn_tensor_load_to_lds(g0, g1, z4, z4, 0);
#endif
        __builtin_amdgcn_s_wait_tensorcnt(0);         // data landed in LDS
    }
    __syncthreads();                                  // publish to all 8 waves

    const int half = lane >> 4;
    const int mr   = lane & 15;
    const int node = geo[m_tile * 16 + mr];           // gather A rows via geo_ids
    const float* arow = A2 + (size_t)node * D_H;

    v8f c0 = {}, c1 = {}, c2 = {}, c3 = {};
#pragma unroll
    for (int kk = 0; kk < G2_KKS; ++kk) {
        v16bf a = load_a_frag(arow, kk * 32, half);
        const __bf16* bp = ldsB + ((size_t)kk * 32 + lane) * 16;
        const size_t bstride = (size_t)G2_KKS * 32 * 16;   // next n_tile in slab
        v16bf b0 = *(const v16bf*)(bp);
        v16bf b1 = *(const v16bf*)(bp + bstride);
        v16bf b2 = *(const v16bf*)(bp + 2 * bstride);
        v16bf b3 = *(const v16bf*)(bp + 3 * bstride);
        c0 = __builtin_amdgcn_wmma_f32_16x16x32_bf16(false, a, false, b0, (short)0, c0, false, false);
        c1 = __builtin_amdgcn_wmma_f32_16x16x32_bf16(false, a, false, b1, (short)0, c1, false, false);
        c2 = __builtin_amdgcn_wmma_f32_16x16x32_bf16(false, a, false, b2, (short)0, c2, false, false);
        c3 = __builtin_amdgcn_wmma_f32_16x16x32_bf16(false, a, false, b3, (short)0, c3, false, false);
    }

    v8f acc[4] = {c0, c1, c2, c3};
#pragma unroll
    for (int j = 0; j < 4; ++j) {
        int n_col = (n_group * 4 + j) * 16 + mr;
        float bb = bias[n_col];
#pragma unroll
        for (int v = 0; v < 8; ++v) {                 // C/D: M = v + 8*half
            int r = m_tile * 16 + (half << 3) + v;
            float val = acc[j][v] + bb;
            val = val > 0.0f ? val : 0.0f;
            out[(size_t)r * D_OUT + n_col] = (__bf16)val;
        }
    }
}

// ---------------- launcher ----------------
extern "C" void kernel_launch(void* const* d_in, const int* in_sizes, int n_in,
                              void* d_out, int out_size, void* d_ws, size_t ws_size,
                              hipStream_t stream) {
    const float* X   = (const float*)d_in[0];
    const float* W1  = (const float*)d_in[1];
    const float* b1  = (const float*)d_in[2];
    const float* W2  = (const float*)d_in[3];
    const float* b2  = (const float*)d_in[4];
    const int*   row = (const int*)d_in[5];
    const int*   col = row + N_EDGES;
    const int*   geo = (const int*)d_in[6];

    // workspace carve-up (256B aligned slices)
    char*  ws  = (char*)d_ws;
    size_t off = 0;
    auto carve = [&](size_t bytes) {
        char* p = ws + off;
        off = (off + bytes + 255) & ~(size_t)255;
        return p;
    };
    float*  dinv = (float*) carve((size_t)N_NODES * 4);
    int*    mask = (int*)   carve((size_t)N_NODES * 4);
    float*  agg1 = (float*) carve((size_t)N_NODES * D_IN * 4);
    __bf16* h1   = (__bf16*)carve((size_t)N_NODES * D_H * 2);
    float*  agg2 = (float*) carve((size_t)N_NODES * D_H * 4);
    __bf16* W1p  = (__bf16*)carve((size_t)D_IN * D_H * 2);
    __bf16* W2p  = (__bf16*)carve((size_t)D_H * D_OUT * 2);
    (void)ws_size; (void)in_sizes; (void)n_in; (void)out_size;

    // weight pre-pack (fragment-ordered bf16)
    {
        int t1 = (D_H / 16) * (D_IN / 32) * 32;      // 1024 threads
        pack_w_kernel<<<(t1 + 255) / 256, 256, 0, stream>>>(W1, W1p, D_IN, D_H);
        int t2 = (D_OUT / 16) * (D_H / 32) * 32;     // 16384 threads
        pack_w_kernel<<<(t2 + 255) / 256, 256, 0, stream>>>(W2, W2p, D_H, D_OUT);
    }

    // degree -> dinv
    deg_init_kernel<<<(N_NODES + 255) / 256, 256, 0, stream>>>(dinv);
    deg_count_kernel<<<(N_EDGES + 255) / 256, 256, 0, stream>>>(col, dinv);
    deg_rsqrt_kernel<<<(N_NODES + 255) / 256, 256, 0, stream>>>(dinv);

    // geo mask
    mask_zero_kernel<<<(N_NODES + 255) / 256, 256, 0, stream>>>(mask);
    mask_set_kernel<<<(N_IDS + 255) / 256, 256, 0, stream>>>(geo, mask);

    // layer 1: aggregate in D_IN=64, then WMMA GEMM -> h1 (bf16)
    agg1_init_kernel<<<(N_NODES * D_IN + 255) / 256, 256, 0, stream>>>(X, dinv, agg1);
    agg1_edge_kernel<<<N_EDGES, D_IN, 0, stream>>>(row, col, dinv, X, agg1);
    gemm1_wmma_kernel<<<(N_NODES / 16 + 1) / 2, 256, 0, stream>>>(agg1, W1p, b1, h1);

    // layer 2: masked aggregate in D_H=256, then TDM+LDS WMMA GEMM -> out
    agg2_init_kernel<<<(N_NODES * D_H + 255) / 256, 256, 0, stream>>>(h1, dinv, mask, agg2);
    agg2_edge_kernel<<<N_EDGES, D_H, 0, stream>>>(row, col, dinv, mask, h1, agg2);
    gemm2_wmma_kernel<<<(N_IDS / 16 / 8) * 16, 256, 0, stream>>>(
        agg2, geo, W2p, b2, (__bf16*)d_out);
}
